// CausalSelfAttention_4054449127959
// MI455X (gfx1250) — matrix operand order repaired and checked by reading, not compiled, not run on hardware
//
#include <hip/hip_runtime.h>
#include <hip/hip_bf16.h>

typedef __attribute__((ext_vector_type(16))) _Float16 v16h;
typedef __attribute__((ext_vector_type(8)))  float    v8f;

#define T_SEQ 1024
#define D_MODEL 1024
#define N_HEADS 16
#define HD 64

// ---------------------------------------------------------------------------
// CDNA5 helpers
// ---------------------------------------------------------------------------
__device__ __forceinline__ v8f wmma16(v16h a, v16h b, v8f c) {
  return __builtin_amdgcn_wmma_f32_16x16x32_f16(
      /*neg_a=*/false, a, /*neg_b=*/false, b,
      /*c_mod=*/(short)0, c, /*reuse_a=*/false, /*reuse_b=*/false);
}

// LDS byte offset (addrspace(3)) of a generic pointer into __shared__ memory
typedef __attribute__((address_space(3))) const void lds_cvoid;
__device__ __forceinline__ uint32_t ldsOff(const void* p) {
  return (uint32_t)(uintptr_t)(lds_cvoid*)p;
}

// Async 16-byte global -> LDS copy (no VGPR data path, tracked by ASYNCcnt)
__device__ __forceinline__ void asyncCopy16B(const _Float16* g, const _Float16* l) {
  asm volatile("global_load_async_to_lds_b128 %0, %1, off"
               :: "v"(ldsOff(l)), "v"(g) : "memory");
}

__device__ __forceinline__ void waitAsync0() {
#if __has_builtin(__builtin_amdgcn_s_wait_asynccnt)
  __builtin_amdgcn_s_wait_asynccnt(0);
#else
  asm volatile("s_wait_asynccnt 0x0" ::: "memory");
#endif
}

// A-matrix 16x32 f16 fragment (ISA 7.12.2): lane&15 = row M,
// lane<16 -> halves 0..7=K0..7, 8..15=K16..23 ; lane>=16 -> K8..15 / K24..31
__device__ __forceinline__ v16h load_frag_a(const _Float16* base, int lda, int lane) {
  const int row = lane & 15;
  const int kb  = (lane >> 4) << 3;
  const _Float16* p = base + (size_t)row * lda + kb;
  v16h a;
#pragma unroll
  for (int i = 0; i < 8; ++i) a[i] = p[i];
#pragma unroll
  for (int i = 0; i < 8; ++i) a[8 + i] = p[16 + i];
  return a;
}

// B-matrix 32x16 f16 fragment, operand stored N-major [n][k]:
// lane&15 = col N; lane<16 -> K0..7, K16..23 ; lane>=16 -> K8..15 / K24..31
__device__ __forceinline__ v16h load_frag_b(const _Float16* base, int ldb, int lane) {
  const int col = lane & 15;
  const int kb  = (lane >> 4) << 3;
  const _Float16* p = base + (size_t)col * ldb + kb;
  v16h b;
#pragma unroll
  for (int i = 0; i < 8; ++i) b[i] = p[i];
#pragma unroll
  for (int i = 0; i < 8; ++i) b[8 + i] = p[16 + i];
  return b;
}

// ---------------------------------------------------------------------------
// Kernel 0: f32 -> f16 conversion (x, w_qkv, w_proj), one pass over HBM
// ---------------------------------------------------------------------------
__global__ void f32_to_f16_kernel(const float* __restrict__ src,
                                  _Float16* __restrict__ dst, int n4) {
  const int i = blockIdx.x * blockDim.x + threadIdx.x;
  if (i < n4) {
    const float4 v = ((const float4*)src)[i];
    _Float16* d = dst + (size_t)i * 4;
    d[0] = (_Float16)v.x; d[1] = (_Float16)v.y;
    d[2] = (_Float16)v.z; d[3] = (_Float16)v.w;
  }
}

// ---------------------------------------------------------------------------
// Kernel 1: HoPE bias table  biasTab[d][h], d = (i-j) + (T-1)
// ---------------------------------------------------------------------------
__global__ void hope_bias_kernel(const float* __restrict__ pos_ind,
                                 float* __restrict__ biasTab) {
  const int idx = blockIdx.x * blockDim.x + threadIdx.x;
  if (idx >= (2 * T_SEQ - 1) * N_HEADS) return;
  const int h = idx & (N_HEADS - 1);
  const int d = idx >> 4;
  const float delta = (float)(d - (T_SEQ - 1));
  float sum = 0.0f;
  if (h < 8) {
#pragma unroll 4
    for (int j = 0; j < 32; ++j) {
      const int p = h * 32 + j;
      const float f  = __powf(10000.0f, -(float)p / 512.0f);
      const float th = delta * f;
      sum += __cosf(th) + __sinf(th);
    }
  } else {
    const int base = (h - 8) * 32;
#pragma unroll 4
    for (int j = 0; j < 32; ++j)
      sum += pos_ind[2 * (base + j)] + pos_ind[2 * (base + j) + 1];
  }
  biasTab[(size_t)d * N_HEADS + h] = sum;
}

// ---------------------------------------------------------------------------
// Kernel 2/4: GEMM  out[M,N] = A[M,K] * W[N,K]^T + bias[N]   (A, W in f16)
// 256 threads = 8 waves (4M x 2N), 128x64 tile, K-tile 64.
// Double-buffered LDS fed by async global->LDS copies (ASYNCcnt pipelined).
// ---------------------------------------------------------------------------
#define GEMM_BUF_HALFS (128 * 72 + 64 * 72)

__device__ __forceinline__ void gemm_issue_tile(
    const _Float16* __restrict__ A, const _Float16* __restrict__ W,
    _Float16* As, _Float16* Bs, int K, int Mbase, int Nbase, int k0, int t) {
#pragma unroll
  for (int i = 0; i < 4; ++i) {  // A tile: 128x64 halfs = 1024 x b128
    const int chunk = t + 256 * i;
    const int row = chunk >> 3, c8 = (chunk & 7) * 8;
    asyncCopy16B(A + (size_t)(Mbase + row) * K + k0 + c8, As + row * 72 + c8);
  }
#pragma unroll
  for (int i = 0; i < 2; ++i) {  // W tile: 64x64 halfs = 512 x b128
    const int chunk = t + 256 * i;
    const int row = chunk >> 3, c8 = (chunk & 7) * 8;
    asyncCopy16B(W + (size_t)(Nbase + row) * K + k0 + c8, Bs + row * 72 + c8);
  }
}

template <bool OUT_HALF>
__global__ __launch_bounds__(256) void gemm_wmma_kernel(
    const _Float16* __restrict__ A, const _Float16* __restrict__ W,
    const float* __restrict__ bias, void* __restrict__ outp,
    int M, int N, int K) {
  __shared__ _Float16 smem[2][GEMM_BUF_HALFS];
  const int t    = threadIdx.x;
  const int w    = t >> 5;
  const int lane = t & 31;
  const int wm   = w >> 1;   // 0..3
  const int wn   = w & 1;    // 0..1
  const int Mbase = blockIdx.y * 128;
  const int Nbase = blockIdx.x * 64;

  v8f acc[2][2] = {};

  const int nk = K / 64;
  gemm_issue_tile(A, W, smem[0], smem[0] + 128 * 72, K, Mbase, Nbase, 0, t);

  for (int kt = 0; kt < nk; ++kt) {
    const int cur = kt & 1;
    waitAsync0();          // this wave's copies for tile kt have landed
    __syncthreads();       // all waves' copies landed; prev compute done
    if (kt + 1 < nk)
      gemm_issue_tile(A, W, smem[cur ^ 1], smem[cur ^ 1] + 128 * 72,
                      K, Mbase, Nbase, (kt + 1) * 64, t);
    const _Float16* As = smem[cur];
    const _Float16* Bs = smem[cur] + 128 * 72;
#pragma unroll
    for (int kk = 0; kk < 64; kk += 32) {
      v16h a0 = load_frag_a(As + (wm * 32) * 72 + kk, 72, lane);
      v16h a1 = load_frag_a(As + (wm * 32 + 16) * 72 + kk, 72, lane);
      v16h b0 = load_frag_b(Bs + (wn * 32) * 72 + kk, 72, lane);
      v16h b1 = load_frag_b(Bs + (wn * 32 + 16) * 72 + kk, 72, lane);
      acc[0][0] = wmma16(a0, b0, acc[0][0]);
      acc[0][1] = wmma16(a0, b1, acc[0][1]);
      acc[1][0] = wmma16(a1, b0, acc[1][0]);
      acc[1][1] = wmma16(a1, b1, acc[1][1]);
    }
    __syncthreads();       // all reads of smem[cur] done before its refill
  }

#pragma unroll
  for (int i = 0; i < 2; ++i)
#pragma unroll
    for (int j = 0; j < 2; ++j) {
      const int colg  = Nbase + wn * 32 + j * 16 + (lane & 15);
      const float bv  = bias[colg];
      const int rowg0 = Mbase + wm * 32 + i * 16 + ((lane >> 4) << 3);
#pragma unroll
      for (int r = 0; r < 8; ++r) {
        const float v = acc[i][j][r] + bv;
        const size_t off = (size_t)(rowg0 + r) * N + colg;
        if (OUT_HALF) ((_Float16*)outp)[off] = (_Float16)v;
        else          ((float*)outp)[off]    = v;
      }
    }
}

// ---------------------------------------------------------------------------
// Kernel 3: causal flash-attention per (q-block, head, batch).
// 128 threads = 4 waves; wave w owns q rows [qbase+16w, qbase+16w+15].
// K tile staged via async global->LDS, V transposed through VGPRs.
// ---------------------------------------------------------------------------
__global__ __launch_bounds__(128) void attn_kernel(
    const _Float16* __restrict__ qkv,   // [B*T][3072]  (q|k|v, head-major)
    const float* __restrict__ biasTab,  // [2T-1][H]
    _Float16* __restrict__ y) {         // [B*T][1024]
  const int qb   = blockIdx.x;   // 0..15
  const int h    = blockIdx.y;   // 0..15
  const int b    = blockIdx.z;   // 0..7
  const int tid  = threadIdx.x;  // 0..127
  const int w    = tid >> 5;
  const int lane = tid & 31;
  const int qbase = qb * 64;

  __shared__ _Float16 Ks[64 * 72];
  __shared__ _Float16 Vt[64 * 72];       // transposed: [hd][key]
  __shared__ _Float16 Ps[4][16 * 72];    // per-wave P staging
  __shared__ float    biasC[T_SEQ];      // bias column for this head

  for (int d = tid; d < T_SEQ; d += 128)
    biasC[d] = biasTab[(size_t)(d + T_SEQ - 1) * N_HEADS + h];

  // Q fragments: persistent in registers, loaded straight from global (f16)
  const _Float16* qrow = qkv + (size_t)(b * T_SEQ + qbase + w * 16) * 3072 + h * HD;
  v16h qf0 = load_frag_a(qrow, 3072, lane);
  v16h qf1 = load_frag_a(qrow + 32, 3072, lane);

  v8f o[4] = {};
  float m[8], l[8];
#pragma unroll
  for (int r = 0; r < 8; ++r) { m[r] = -1e30f; l[r] = 0.0f; }

  const int nj = qb + 1;  // causal: key tiles 0..qb
  for (int j = 0; j < nj; ++j) {
    const int jb = j * 64;
    __syncthreads();  // previous iteration's Ks/Vt reads complete

    // K tile [key][hd]: async global->LDS, 4 x b128 per thread (overlaps below)
#pragma unroll
    for (int i = 0; i < 4; ++i) {
      const int chunk = tid + 128 * i;            // 0..511
      const int row = chunk >> 3, c8 = (chunk & 7) * 8;
      asyncCopy16B(qkv + (size_t)(b * T_SEQ + jb + row) * 3072 + D_MODEL + h * HD + c8,
                   Ks + row * 72 + c8);
    }
    // V tile transposed [hd][key] through VGPRs while K streams in
    {
      const int row = tid >> 1, cb = (tid & 1) * 32;
      const _Float16* vsrc =
          qkv + (size_t)(b * T_SEQ + jb + row) * 3072 + 2 * D_MODEL + h * HD + cb;
#pragma unroll
      for (int i = 0; i < 32; ++i)
        Vt[(cb + i) * 72 + row] = vsrc[i];
    }
    waitAsync0();
    __syncthreads();

    // S = Q * K^T  (4 col-subtiles of 16)
    v8f s[4] = {};
#pragma unroll
    for (int sub = 0; sub < 4; ++sub) {
      v16h b0 = load_frag_b(Ks + (sub * 16) * 72, 72, lane);
      v16h b1 = load_frag_b(Ks + (sub * 16) * 72 + 32, 72, lane);
      s[sub] = wmma16(qf0, b0, s[sub]);
      s[sub] = wmma16(qf1, b1, s[sub]);
    }

    // online softmax (rows span 16-lane halves of the wave)
    float pr[4][8];
#pragma unroll
    for (int r = 0; r < 8; ++r) {
      const int q = qbase + w * 16 + ((lane >> 4) << 3) + r;
      float rowmax = -1e30f;
#pragma unroll
      for (int sub = 0; sub < 4; ++sub) {
        const int k = jb + sub * 16 + (lane & 15);
        const int d = q - k;
        float v = (d < 0) ? -1e30f : (s[sub][r] * 0.125f + biasC[d]);
        pr[sub][r] = v;
        rowmax = fmaxf(rowmax, v);
      }
#pragma unroll
      for (int off = 1; off < 16; off <<= 1)
        rowmax = fmaxf(rowmax, __shfl_xor(rowmax, off, 32));
      const float mn = fmaxf(m[r], rowmax);
      const float alpha = __expf(m[r] - mn);
      m[r] = mn;
      float rs = 0.0f;
#pragma unroll
      for (int sub = 0; sub < 4; ++sub) {
        const float p = __expf(pr[sub][r] - mn);
        pr[sub][r] = p;
        rs += p;
      }
#pragma unroll
      for (int off = 1; off < 16; off <<= 1)
        rs += __shfl_xor(rs, off, 32);
      l[r] = l[r] * alpha + rs;
#pragma unroll
      for (int sub = 0; sub < 4; ++sub) o[sub][r] *= alpha;
    }

    // stage P (C-layout -> row-major f16, per-wave region; same-wave LDS
    // operations are processed in order)
    _Float16* P = &Ps[w][0];
#pragma unroll
    for (int r = 0; r < 8; ++r) {
      const int prow = ((lane >> 4) << 3) + r;
#pragma unroll
      for (int sub = 0; sub < 4; ++sub)
        P[prow * 72 + sub * 16 + (lane & 15)] = (_Float16)pr[sub][r];
    }

    // O += P * V
#pragma unroll
    for (int ks = 0; ks < 2; ++ks) {
      v16h ap = load_frag_a(P + ks * 32, 72, lane);
#pragma unroll
      for (int sub = 0; sub < 4; ++sub) {
        v16h bv = load_frag_b(Vt + (sub * 16) * 72 + ks * 32, 72, lane);
        o[sub] = wmma16(ap, bv, o[sub]);
      }
    }
  }

  // epilogue: normalize and store f16 (layout [b,t,h*64+d] == reference merge)
#pragma unroll
  for (int r = 0; r < 8; ++r) {
    const int q = qbase + w * 16 + ((lane >> 4) << 3) + r;
    const float inv = 1.0f / l[r];
    const size_t base = ((size_t)(b * T_SEQ) + q) * D_MODEL + h * HD;
#pragma unroll
    for (int sub = 0; sub < 4; ++sub)
      y[base + sub * 16 + (lane & 15)] = (_Float16)(o[sub][r] * inv);
  }
}

// ---------------------------------------------------------------------------
extern "C" void kernel_launch(void* const* d_in, const int* in_sizes, int n_in,
                              void* d_out, int out_size, void* d_ws, size_t ws_size,
                              hipStream_t stream) {
  const float* x        = (const float*)d_in[0];  // [8,1024,1024]
  const float* w_qkv    = (const float*)d_in[1];  // [3072,1024]
  const float* b_qkv    = (const float*)d_in[2];  // [3072]
  const float* w_proj   = (const float*)d_in[3];  // [1024,1024]
  const float* b_proj   = (const float*)d_in[4];  // [1024]
  const float* pos_ind  = (const float*)d_in[5];  // [256,2]
  float* out = (float*)d_out;                     // [8,1024,1024]

  const int M = 8 * T_SEQ;  // 8192

  // workspace layout (bytes)
  char* ws = (char*)d_ws;
  size_t off = 0;
  float*    biasTab = (float*)(ws + off);    off += (size_t)1 << 17;           // 128KB
  _Float16* xh      = (_Float16*)(ws + off); off += (size_t)M * D_MODEL * 2;   // 16MB
  _Float16* wqh     = (_Float16*)(ws + off); off += (size_t)3 * D_MODEL * D_MODEL * 2; // 6MB
  _Float16* wph     = (_Float16*)(ws + off); off += (size_t)D_MODEL * D_MODEL * 2;     // 2MB
  _Float16* qkv     = (_Float16*)(ws + off); off += (size_t)M * 3 * D_MODEL * 2;       // 48MB
  _Float16* y       = (_Float16*)(ws + off);                                           // 16MB

  // 0) downconvert operands once (single HBM pass)
  {
    const int nx = M * D_MODEL / 4, nq = 3 * D_MODEL * D_MODEL / 4,
              np = D_MODEL * D_MODEL / 4;
    f32_to_f16_kernel<<<(nx + 255) / 256, 256, 0, stream>>>(x, xh, nx);
    f32_to_f16_kernel<<<(nq + 255) / 256, 256, 0, stream>>>(w_qkv, wqh, nq);
    f32_to_f16_kernel<<<(np + 255) / 256, 256, 0, stream>>>(w_proj, wph, np);
  }

  // 1) HoPE bias table
  hope_bias_kernel<<<((2 * T_SEQ - 1) * N_HEADS + 255) / 256, 256, 0, stream>>>(
      pos_ind, biasTab);

  // 2) QKV projection: [8192x1024] x [3072x1024]^T -> f16
  gemm_wmma_kernel<true><<<dim3(3 * D_MODEL / 64, M / 128), 256, 0, stream>>>(
      xh, wqh, b_qkv, (void*)qkv, M, 3 * D_MODEL, D_MODEL);

  // 3) causal flash-attention with HoPE bias
  attn_kernel<<<dim3(T_SEQ / 64, N_HEADS, 8), 128, 0, stream>>>(qkv, biasTab, y);

  // 4) output projection: [8192x1024] x [1024x1024]^T -> f32 out
  gemm_wmma_kernel<false><<<dim3(D_MODEL / 64, M / 128), 256, 0, stream>>>(
      y, wph, b_proj, (void*)out, M, D_MODEL, D_MODEL);
}
